// E2RGATLoss_20959440405252
// MI455X (gfx1250) — matrix operand order repaired and checked by reading, not compiled, not run on hardware
//
#include <hip/hip_runtime.h>
#include <hip/hip_bf16.h>

// CDNA5 / gfx1250 WMMA operand types
typedef __attribute__((ext_vector_type(16))) __bf16 v16bf;
typedef __attribute__((ext_vector_type(8)))  float  v8f;

#define MT 4  // row-tiles (of 16 anchors) per wave in k_main

__device__ __forceinline__ float wred(float v) {
#pragma unroll
  for (int m = 16; m >= 1; m >>= 1) v += __shfl_xor(v, m, 32);
  return v;
}

__global__ void k_zero(float* acc) { acc[0] = 0.0f; }

// Block roles:
//  [0, pairBlocks)            : one wave per pair -> bf16 anchor (scaled by
//                               log2e/T) + pos_sim/T (f32)
//  [pairBlocks, +negBlocks)   : one wave per negative -> bf16 normalized row
//  last block                 : zero the padded anchor rows [P, Ppad)
__global__ void k_prep(const float* __restrict__ h,
                       const int* __restrict__ pos_pairs,
                       const int* __restrict__ neg_pairs,
                       const float* __restrict__ temp,
                       float* __restrict__ pos_ws,
                       __hip_bfloat16* __restrict__ nn_ws,
                       __hip_bfloat16* __restrict__ an_ws,
                       int P, int K, int Ppad, int pairBlocks) {
  const int lane = threadIdx.x & 31;
  const int wv = threadIdx.x >> 5;
  const float invT = 1.0f / temp[0];
  if ((int)blockIdx.x < pairBlocks) {
    const int p = (int)blockIdx.x * 8 + wv;
    if (p >= P) return;
    const int ia = pos_pairs[p];
    const int ip = pos_pairs[P + p];
    const float* ha = h + (size_t)ia * 64;
    const float* hp = h + (size_t)ip * 64;
    const float a0 = ha[lane], a1 = ha[lane + 32];
    const float q0 = hp[lane], q1 = hp[lane + 32];
    const float ssa = wred(a0 * a0 + a1 * a1);
    const float ssp = wred(q0 * q0 + q1 * q1);
    const float dp  = wred(a0 * q0 + a1 * q1);
    const float inva = 1.0f / fmaxf(sqrtf(ssa), 1e-8f);
    const float invp = 1.0f / fmaxf(sqrtf(ssp), 1e-8f);
    if (lane == 0) pos_ws[p] = dp * inva * invp * invT;  // natural sim/T
    // fold 1/T AND log2(e) into A: WMMA result is directly exp2-domain logits
    const float sA = inva * invT * 1.44269504088896340736f;
    an_ws[(size_t)p * 64 + lane]      = __float2bfloat16(a0 * sA);
    an_ws[(size_t)p * 64 + lane + 32] = __float2bfloat16(a1 * sA);
  } else if ((int)blockIdx.x < pairBlocks + (K + 7) / 8) {
    const int k = ((int)blockIdx.x - pairBlocks) * 8 + wv;
    if (k >= K) return;
    const int in_ = neg_pairs[K + k];  // neg_pairs[1][k]
    const float* hn = h + (size_t)in_ * 64;
    const float n0 = hn[lane], n1 = hn[lane + 32];
    const float ss = wred(n0 * n0 + n1 * n1);
    const float inv = 1.0f / fmaxf(sqrtf(ss), 1e-8f);
    nn_ws[(size_t)k * 64 + lane]      = __float2bfloat16(n0 * inv);
    nn_ws[(size_t)k * 64 + lane + 32] = __float2bfloat16(n1 * inv);
  } else {
    // zero padded anchor rows so padded tiles produce harmless logits (2^0)
    uint4* pad = (uint4*)(an_ws + (size_t)P * 64);
    const int nq = (Ppad - P) * 8;  // 64 bf16 = 8 uint4 per row
    const uint4 z = make_uint4(0u, 0u, 0u, 0u);
    for (int i = threadIdx.x; i < nq; i += blockDim.x) pad[i] = z;
  }
}

// Branch-free GEMM (bf16 WMMA 16x16x32, F=64 => 2 chained WMMAs per 16x16
// tile) + fused logsumexp. Negatives in LDS (64KB), rotating (off+row)&7 chunk swizzle.
// Each wave owns MT=4 row-tiles (64 anchors): every B-fragment ds_load feeds
// 4 WMMA-pairs; inner loop runs with full EXEC, no divergence.
__global__ void __launch_bounds__(128)
k_main(const uint4* __restrict__ nn_q,
       const __hip_bfloat16* __restrict__ an_ws,
       const float* __restrict__ pos_ws,
       float* __restrict__ acc,
       int P, int K, int Ppad) {
  __shared__ uint4 smem[4096];  // up to 512 rows x 8 chunks x 16B = 64KB
  const int nch = K * 8;
  for (int c = threadIdx.x; c < nch; c += blockDim.x) {
    const int row = c >> 3, off = c & 7;
    smem[(row << 3) | ((off + row) & 7)] = nn_q[c];
  }
  __syncthreads();

  const int lane = threadIdx.x & 31;
  const int wv = threadIdx.x >> 5;
  const int l15 = lane & 15, half = lane >> 4;
  const int ntilesPad = Ppad >> 4;
  const int tile0 = ((int)blockIdx.x * 4 + wv) * MT;
  if (tile0 >= ntilesPad) return;  // wave-uniform

  union U { uint4 q[2]; v16bf v; };
  U a0u[MT], a1u[MT];
  float s[MT][8];
#pragma unroll
  for (int m = 0; m < MT; ++m) {
    // 16-bit A 16x32 layout: lanes 0-15 hold K{0-7,16-23}, lanes 16-31 K{8-15,24-31}
    const uint4* pa =
        (const uint4*)(an_ws + (size_t)(tile0 + m) * 16 * 64 + (size_t)l15 * 64) + half;
    a0u[m].q[0] = pa[0]; a0u[m].q[1] = pa[2];  // features 0..31
    a1u[m].q[0] = pa[4]; a1u[m].q[1] = pa[6];  // features 32..63
#pragma unroll
    for (int i = 0; i < 8; ++i) s[m][i] = 0.0f;
  }

  const int NT = K >> 4;
  const int c0 = half;
  for (int t = 0; t < NT; ++t) {
    const int nrow = (t << 4) + l15;   // this lane's negative (column)
    const int base = nrow << 3;
    // 16-bit B 32x16 layout: lane = column, lanes 0-15 hold K{0-7,16-23}, lanes 16-31 K{8-15,24-31}
    U b0u, b1u;
    b0u.q[0] = smem[base | ((c0     + nrow) & 7)];
    b0u.q[1] = smem[base | ((c0 + 2 + nrow) & 7)];
    b1u.q[0] = smem[base | ((c0 + 4 + nrow) & 7)];
    b1u.q[1] = smem[base | ((c0 + 6 + nrow) & 7)];
#pragma unroll
    for (int m = 0; m < MT; ++m) {
      v8f cacc = {};
      cacc = __builtin_amdgcn_wmma_f32_16x16x32_bf16(
          false, a0u[m].v, false, b0u.v, (short)0, cacc, false, false);
      cacc = __builtin_amdgcn_wmma_f32_16x16x32_bf16(
          false, a1u[m].v, false, b1u.v, (short)0, cacc, false, false);
      // logits already in exp2 domain (log2e folded into A); bounded by
      // |cos|*log2e/T so no max-subtraction is needed for stability
#pragma unroll
      for (int i = 0; i < 8; ++i) s[m][i] += __builtin_amdgcn_exp2f(cacc[i]);
    }
  }

  const int ntiles = P >> 4;
#pragma unroll
  for (int m = 0; m < MT; ++m) {
#pragma unroll
    for (int i = 0; i < 8; ++i) {
#pragma unroll
      for (int d = 1; d < 16; d <<= 1) s[m][i] += __shfl_xor(s[m][i], d, 32);
    }
    const int tile = tile0 + m;
    if (tile < ntiles && l15 == 0) {  // lanes 0 & 16: rows 0-7 / 8-15
      const int rbase = tile * 16 + half * 8;
      float part = 0.0f;
#pragma unroll
      for (int i = 0; i < 8; ++i) {
        const float pv = pos_ws[rbase + i];
        part += __logf(s[m][i] + __expf(pv)) - pv;
      }
      atomicAdd(acc, part);
    }
  }
}

__global__ void k_final(const float* __restrict__ acc, float* __restrict__ out,
                        float invP) {
  out[0] = acc[0] * invP;
}

extern "C" void kernel_launch(void* const* d_in, const int* in_sizes, int n_in,
                              void* d_out, int out_size, void* d_ws, size_t ws_size,
                              hipStream_t stream) {
  const float* h         = (const float*)d_in[0];
  const int*   pos_pairs = (const int*)d_in[1];   // (2,P) flat
  const int*   neg_pairs = (const int*)d_in[2];   // (2,K) flat
  // d_in[3] relation_types: unused (partitions sum only)
  const float* temp      = (const float*)d_in[4];

  const int P = in_sizes[1] / 2;
  const int K = in_sizes[2] / 2;
  const int ntiles    = (P + 15) / 16;
  const int ntilesPad = ((ntiles + MT - 1) / MT) * MT;
  const int Ppad      = ntilesPad * 16;

  char* ws = (char*)d_ws;
  float* acc    = (float*)ws;                                   // 1 float
  float* pos_ws = (float*)(ws + 256);                           // P floats
  size_t nn_off = (256 + (size_t)P * 4 + 255) & ~(size_t)255;
  __hip_bfloat16* nn_ws = (__hip_bfloat16*)(ws + nn_off);       // K*64 bf16
  size_t an_off = (nn_off + (size_t)K * 64 * 2 + 255) & ~(size_t)255;
  __hip_bfloat16* an_ws = (__hip_bfloat16*)(ws + an_off);       // Ppad*64 bf16

  k_zero<<<1, 1, 0, stream>>>(acc);

  const int pairBlocks = (P + 7) / 8;
  const int negBlocks  = (K + 7) / 8;
  k_prep<<<pairBlocks + negBlocks + 1, 256, 0, stream>>>(
      h, pos_pairs, neg_pairs, temp, pos_ws, nn_ws, an_ws, P, K, Ppad, pairBlocks);

  const int jobs   = ntilesPad / MT;         // waves needed
  const int blocks = (jobs + 3) / 4;         // 4 waves per block
  k_main<<<blocks, 128, 0, stream>>>((const uint4*)nn_ws, an_ws, pos_ws, acc,
                                     P, K, Ppad);

  k_final<<<1, 1, 0, stream>>>(acc, (float*)d_out, 1.0f / (float)P);
}
